// ViTAttention_49967649522063
// MI455X (gfx1250) — matrix operand order repaired and checked by reading, not compiled, not run on hardware
//
#include <hip/hip_runtime.h>

typedef __attribute__((ext_vector_type(16))) _Float16 v16h;
typedef __attribute__((ext_vector_type(8)))  _Float16 v8h;
typedef __attribute__((ext_vector_type(4)))  _Float16 v4h;
typedef __attribute__((ext_vector_type(8)))  float    v8f;
typedef __attribute__((ext_vector_type(4)))  float    v4f;
typedef __attribute__((ext_vector_type(4)))  int      v4i;

constexpr int B  = 16;
constexpr int S  = 577;
constexpr int H  = 1024;
constexpr int NH = 16;
constexpr int HD = 64;
constexpr int M  = B * S;     // 9232 (multiple of 16)
constexpr int N3 = 3 * H;     // 3072 fused QKV output dim
constexpr int SP = 608;       // padded seq len (mult of 32, >= 577+31)
constexpr int QT = 37;        // ceil(S/16) query tiles
constexpr int QTB = 5;        // q-tile blocks per (b,h): ceil(37/8)
constexpr int NKT = 19;       // key tiles of 32: ceil(577/32)

#if defined(__AMDGCN__) && __has_builtin(__builtin_amdgcn_global_load_async_to_lds_b128)
#define USE_ASYNC_LDS 1
#else
#define USE_ASYNC_LDS 0
#endif

static __device__ __forceinline__ v16h cat16(v8h lo, v8h hi) {
  return __builtin_shufflevector(lo, hi, 0,1,2,3,4,5,6,7,8,9,10,11,12,13,14,15);
}
static __device__ __forceinline__ v8f wmma16(v16h a, v16h b, v8f c) {
  return __builtin_amdgcn_wmma_f32_16x16x32_f16(false, a, false, b, (short)0, c,
                                                false, false);
}
// 16B global -> LDS copy (async on CDNA5 if available, else via VGPRs)
static __device__ __forceinline__ void copy16_to_lds(const _Float16* g,
                                                     _Float16* l) {
#if USE_ASYNC_LDS
  __builtin_amdgcn_global_load_async_to_lds_b128(
      (__attribute__((address_space(1))) v4i*)g,
      (__attribute__((address_space(3))) v4i*)l, 0, 0);
#else
  *(v8h*)l = *(const v8h*)g;
#endif
}
static __device__ __forceinline__ void wait_async_copies() {
#if USE_ASYNC_LDS
#if __has_builtin(__builtin_amdgcn_s_wait_asynccnt)
  __builtin_amdgcn_s_wait_asynccnt(0);
#else
  asm volatile("s_wait_asynccnt 0" ::: "memory");
#endif
#endif
}

// ---------------------------------------------------------------- cvt f32->f16
__global__ __launch_bounds__(256) void cvt_kernel(
    const float* __restrict__ x,  const float* __restrict__ wq,
    const float* __restrict__ wk, const float* __restrict__ wv,
    _Float16* __restrict__ xh, _Float16* __restrict__ wh) {
  const long NX = (long)M * H / 4;
  const long NW = (long)H * H / 4;
  long i = (long)blockIdx.x * blockDim.x + threadIdx.x;
  if (i < NX) {
    v4f v = ((const v4f*)x)[i];
    ((v4h*)xh)[i] = __builtin_convertvector(v, v4h);
  } else if (i < NX + 3 * NW) {
    long j = i - NX;
    const float* src = (j < NW) ? wq : (j < 2 * NW ? wk : wv);
    v4f v = ((const v4f*)src)[j % NW];
    ((v4h*)wh)[j] = __builtin_convertvector(v, v4h);
  }
}

// ---------------------------------------------------------------- zero pad buf
__global__ __launch_bounds__(256) void zero_kernel(_Float16* __restrict__ p,
                                                   long n8) {
  long i = (long)blockIdx.x * blockDim.x + threadIdx.x;
  if (i < n8) { v8h z = {}; ((v8h*)p)[i] = z; }
}

// ---------------------------------------------------------------- fused QKV GEMM
// Y[m,n] = sum_k X[m,k]*Wall[n,k] + b[n].  Block tile 128(M) x 256(N),
// 8 waves as 2(M) x 4(N); wave tile 64x64 = 4x4 WMMA frags (16 WMMAs / 8 loads).
__global__ __launch_bounds__(256) void gemm_qkv_kernel(
    const _Float16* __restrict__ Xh, const _Float16* __restrict__ Wh,
    const float* __restrict__ bq, const float* __restrict__ bk,
    const float* __restrict__ bv,
    _Float16* __restrict__ Q, _Float16* __restrict__ Kb,
    _Float16* __restrict__ Vt) {
  const int lane = threadIdx.x & 31;
  const int w    = threadIdx.x >> 5;
  const int wm   = w & 1;        // 0..1 -> M sub-tile
  const int wn   = w >> 1;       // 0..3 -> N sub-tile
  const int lm   = lane & 15;
  const int hi   = lane >> 4;    // lane half selects K chunk
  const int m0   = blockIdx.y * 128 + wm * 64;
  const int n0   = blockIdx.x * 256 + wn * 64;

  int ar[4];
#pragma unroll
  for (int i = 0; i < 4; ++i) {
    ar[i] = m0 + i * 16 + lm;
    if (ar[i] > M - 1) ar[i] = M - 1;
  }

  v8f acc[4][4] = {};
  const int oA = hi ? 16 : 0;    // A: K 0..15 lo-half, 16..31 hi-half
  const int oB = hi ? 16 : 0;    // B: K 0..15 lo-half, 16..31 hi-half

  for (int kb = 0; kb < H; kb += 32) {
    v16h a[4];
#pragma unroll
    for (int i = 0; i < 4; ++i) {
      const _Float16* pr = Xh + (long)ar[i] * H + kb + oA;
      a[i] = *(const v16h*)pr;
    }
    v16h bf[4];
#pragma unroll
    for (int j = 0; j < 4; ++j) {
      const int n = n0 + j * 16 + lm;
      bf[j] = *(const v16h*)(Wh + (long)n * H + kb + oB);
    }
#pragma unroll
    for (int i = 0; i < 4; ++i)
#pragma unroll
      for (int j = 0; j < 4; ++j)
        acc[i][j] = wmma16(a[i], bf[j], acc[i][j]);
  }

  // epilogue: bias + f16 scatter into Q[b,h,s,d], K[b,h,s,d], Vt[b,h,d,s]
#pragma unroll
  for (int j = 0; j < 4; ++j) {
    const int ng = n0 + j * 16 + lm;              // 0..3071
    const float* bp = (ng < 1024) ? bq : (ng < 2048 ? bk : bv);
    const float bias = bp[ng & 1023];
    const int which = ng >> 10;                   // 0=Q 1=K 2=V
    const int f = ng & 1023;
    const int hh = f >> 6, d = f & 63;
#pragma unroll
    for (int i = 0; i < 4; ++i) {
#pragma unroll
      for (int g = 0; g < 8; ++g) {
        const int m = m0 + i * 16 + g + hi * 8;   // C/D: row = g + 8*hi
        if (m < M) {
          const int b = m / S, s = m - b * S;
          const _Float16 hv = (_Float16)(acc[i][j][g] + bias);
          const long base = (long)(b * NH + hh);
          if (which == 0)      Q [(base * SP + s) * HD + d] = hv;
          else if (which == 1) Kb[(base * SP + s) * HD + d] = hv;
          else                 Vt[(base * HD + d) * SP + s] = hv;
        }
      }
    }
  }
}

// ---------------------------------------------------------------- flash attention
// One block (8 waves) = 8 query tiles of the SAME (b,h); K/V key-tiles staged
// in LDS (double-buffered, async) and shared by all waves.
__global__ __launch_bounds__(256) void attn_kernel(
    const _Float16* __restrict__ Q, const _Float16* __restrict__ Kb,
    const _Float16* __restrict__ Vt, float* __restrict__ out) {
  __shared__ __align__(32) _Float16 ktile[2][32 * 64];  // [key][d]
  __shared__ __align__(32) _Float16 vtile[2][64 * 32];  // [d][key]
  __shared__ __align__(16) float    plds[8 * 16 * 32];  // per-wave P staging

  const int tid  = threadIdx.x;
  const int lane = tid & 31;
  const int w    = tid >> 5;
  const int lm   = lane & 15;
  const int hi   = lane >> 4;
  const int bh   = blockIdx.x / QTB;                 // b*NH + h
  const int blk  = blockIdx.x % QTB;
  const int qt   = blk * 8 + w;                      // 0..39
  const bool qvalid = qt < QT;
  const int qb   = (qvalid ? qt : QT - 1) * 16;
  const long qkBase = (long)bh * SP * HD;
  const long vBase  = (long)bh * HD * SP;

  // cooperative tile loader: K tile 32x64 (8 thr/row), V tile 64x32 (4 thr/row)
  const int kRow = tid >> 3, kCol = (tid & 7) * 8;
  const int vRow = tid >> 2, vCol = (tid & 3) * 8;

  // Q A-fragments for d-steps 0 and 32 (pad rows are zero)
  v16h aq[2];
  {
    const _Float16* pr = Q + qkBase + (long)(qb + lm) * HD + (hi ? 16 : 0);
#pragma unroll
    for (int t = 0; t < 2; ++t)
      aq[t] = *(const v16h*)(pr + t * 32);
  }

  v8f acc[4] = {};
  float mrow[8], lrow[8];
#pragma unroll
  for (int g = 0; g < 8; ++g) { mrow[g] = -3.0e38f; lrow[g] = 0.0f; }

  float* myp = plds + w * (16 * 32);
  const float LOG2E = 1.44269504f;
  const float SCALE = 0.03125f;                     // 1/sqrt(H) = 1/32 exactly

  // prologue: stage first key tile
  copy16_to_lds(Kb + qkBase + (long)kRow * HD + kCol, &ktile[0][kRow * 64 + kCol]);
  copy16_to_lds(Vt + vBase + (long)vRow * SP + vCol, &vtile[0][vRow * 32 + vCol]);
  wait_async_copies();
  __syncthreads();

  for (int it = 0; it < NKT; ++it) {
    const int kb  = it * 32;
    const int cur = it & 1;
    if (it + 1 < NKT) {                              // prefetch next tile
      const int nb = kb + 32, nx = cur ^ 1;
      copy16_to_lds(Kb + qkBase + (long)(nb + kRow) * HD + kCol,
                    &ktile[nx][kRow * 64 + kCol]);
      copy16_to_lds(Vt + vBase + (long)vRow * SP + nb + vCol,
                    &vtile[nx][vRow * 32 + vCol]);
    }
    // ---- scores: 16(q) x 32(k) from LDS K tile
    v8f sf[2];
#pragma unroll
    for (int nt = 0; nt < 2; ++nt) {
      const _Float16* pk = &ktile[cur][(nt * 16 + lm) * 64 + (hi ? 16 : 0)];
      v8f c = {};
      c = wmma16(aq[0], *(const v16h*)pk, c);
      c = wmma16(aq[1], *(const v16h*)(pk + 32), c);
      sf[nt] = c;
    }
    // scale + column mask
#pragma unroll
    for (int nt = 0; nt < 2; ++nt) {
      const bool valid = (kb + nt * 16 + lm) < S;
#pragma unroll
      for (int g = 0; g < 8; ++g)
        sf[nt][g] = valid ? sf[nt][g] * SCALE : -3.0e38f;
    }
    // ---- online softmax (row = g + 8*hi, replicated across 16-lane half)
#pragma unroll
    for (int g = 0; g < 8; ++g) {
      float t = fmaxf(sf[0][g], sf[1][g]);
      t = fmaxf(t, __shfl_xor(t, 1, 32));
      t = fmaxf(t, __shfl_xor(t, 2, 32));
      t = fmaxf(t, __shfl_xor(t, 4, 32));
      t = fmaxf(t, __shfl_xor(t, 8, 32));
      const float mnew = fmaxf(mrow[g], t);
      const float corr = __builtin_amdgcn_exp2f((mrow[g] - mnew) * LOG2E);
      mrow[g] = mnew;
      const float p0 = __builtin_amdgcn_exp2f((sf[0][g] - mnew) * LOG2E);
      const float p1 = __builtin_amdgcn_exp2f((sf[1][g] - mnew) * LOG2E);
      sf[0][g] = p0; sf[1][g] = p1;
      float rs = p0 + p1;
      rs += __shfl_xor(rs, 1, 32);
      rs += __shfl_xor(rs, 2, 32);
      rs += __shfl_xor(rs, 4, 32);
      rs += __shfl_xor(rs, 8, 32);
      lrow[g] = lrow[g] * corr + rs;
#pragma unroll
      for (int j = 0; j < 4; ++j) acc[j][g] *= corr;
    }
    // ---- redistribute P (C/D layout) -> A layout via per-wave LDS region
#pragma unroll
    for (int g = 0; g < 8; ++g) {
      const int r = g + hi * 8;
      myp[r * 32 + lm]      = sf[0][g];
      myp[r * 32 + 16 + lm] = sf[1][g];
    }
    v16h ap = {};
    {
      const v4f* pr4 = (const v4f*)(myp + lm * 32 + (hi ? 8 : 0));
      const v4f c0 = pr4[0], c1 = pr4[1];  // cols {0..7 | 8..15}
      const v4f c2 = pr4[4], c3 = pr4[5];  // cols {16..23 | 24..31}
#pragma unroll
      for (int e = 0; e < 4; ++e) {
        ap[e]      = (_Float16)c0[e];
        ap[4 + e]  = (_Float16)c1[e];
        ap[8 + e]  = (_Float16)c2[e];
        ap[12 + e] = (_Float16)c3[e];
      }
    }
    // ---- ctx += P(16x32) @ V(32x64) from LDS V tile
#pragma unroll
    for (int j = 0; j < 4; ++j) {
      const _Float16* pv = &vtile[cur][(j * 16 + lm) * 32 + (hi ? 8 : 0)];
      acc[j] = wmma16(ap, cat16(*(const v8h*)pv, *(const v8h*)(pv + 16)), acc[j]);
    }
    wait_async_copies();   // next tile resident before anyone reads it
    __syncthreads();       // and everyone done with current tile
  }

  // ---- normalize + scatter to out[b, s, h*64+d] (f32)
  if (qvalid) {
    const int b = bh >> 4, hh = bh & 15;
#pragma unroll
    for (int g = 0; g < 8; ++g) {
      const int s = qb + g + hi * 8;
      if (s < S) {
        const float inv = 1.0f / lrow[g];
#pragma unroll
        for (int j = 0; j < 4; ++j) {
          const int d = j * 16 + lm;
          out[((long)(b * S + s)) * H + hh * HD + d] = acc[j][g] * inv;
        }
      }
    }
  }
}

// ---------------------------------------------------------------- launcher
extern "C" void kernel_launch(void* const* d_in, const int* in_sizes, int n_in,
                              void* d_out, int out_size, void* d_ws,
                              size_t ws_size, hipStream_t stream) {
  const float* x  = (const float*)d_in[0];
  const float* wq = (const float*)d_in[1];
  const float* bq = (const float*)d_in[2];
  const float* wk = (const float*)d_in[3];
  const float* bk = (const float*)d_in[4];
  const float* wv = (const float*)d_in[5];
  const float* bv = (const float*)d_in[6];
  float* out = (float*)d_out;

  _Float16* Xh = (_Float16*)d_ws;                 // M*H          f16
  _Float16* Wh = Xh + (long)M * H;                // 3*H*H        f16
  _Float16* Qb = Wh + 3L * H * H;                 // B*NH*SP*HD   f16
  const long QKVsz = (long)B * NH * SP * HD;      // 9,961,472 halves
  _Float16* Kb = Qb + QKVsz;
  _Float16* Vb = Kb + QKVsz;                      // transposed [b,h,d,s]

  {  // f32 -> f16 conversion of x and fused weights
    const long groups = ((long)M * H + 3L * H * H) / 4;
    const int blocks = (int)((groups + 255) / 256);
    cvt_kernel<<<blocks, 256, 0, stream>>>(x, wq, wk, wv, Xh, Wh);
  }
  {  // zero padded Q/K/Vt (contiguous region)
    const long n8 = 3 * QKVsz / 8;
    const int blocks = (int)((n8 + 255) / 256);
    zero_kernel<<<blocks, 256, 0, stream>>>(Qb, n8);
  }
  {  // fused QKV projection GEMM: M=9232, N=3072, K=1024
    dim3 grid(N3 / 256, (M + 127) / 128);         // 12 x 73
    gemm_qkv_kernel<<<grid, 256, 0, stream>>>(Xh, Wh, bq, bk, bv, Qb, Kb, Vb);
  }
  {  // flash attention: 8 q-tiles per block sharing LDS-staged K/V
    const int blocks = B * NH * QTB;              // 1280 blocks
    attn_kernel<<<blocks, 256, 0, stream>>>(Qb, Kb, Vb, out);
  }
}
